// AttentionBlock_self_7404523618827
// MI455X (gfx1250) — matrix operand order repaired and checked by reading, compiled, not run on hardware
//
#include <hip/hip_runtime.h>

// ---------------------------------------------------------------------------
// AttentionBlock for MI455X (gfx1250, wave32, WMMA).
// B=8, C=512, T=1024, heads=8, ch=64, groups=32.
// All GEMMs: v_wmma_f32_16x16x32_f16 (f16 in, f32 acc).
// Weights pre-converted to f16 once; each GEMM wave register-blocks 16x128.
// Attention keeps exp(scores) as f16 in LDS (128KB/block -> 2 blocks/WGP on
// CDNA5's 320KB LDS); softmax 1/sum is folded into the P*V^T epilogue.
// ---------------------------------------------------------------------------

typedef _Float16 v16h __attribute__((ext_vector_type(16)));
typedef _Float16 v8h  __attribute__((ext_vector_type(8)));
typedef _Float16 v2h  __attribute__((ext_vector_type(2)));
typedef float    v8f  __attribute__((ext_vector_type(8)));
typedef float    v4f  __attribute__((ext_vector_type(4)));

#define NB   8
#define NC   512
#define NT   1024
#define NHD  8
#define NCH  64
#define NBH  (NB * NHD)                 // 64
#define QK_SCALE 0.35355339059327373f   // 64^(-0.25)

// ---------------- WMMA fragment loaders (per CDNA5 ISA VGPR layouts) -------

// A (16x32 f16, row-major, ld): lane l<16 -> row m=l, K={0..7,16..23};
// lanes 16..31 -> K={8..15,24..31}.
static __device__ __forceinline__ v16h fragA_f16(const _Float16* p, int ld, int lane) {
  int row = lane & 15;
  int ko  = (lane >> 4) << 3;
  const _Float16* q = p + row * ld + ko;
  v8h lo = *(const v8h*)(q);
  v8h hi = *(const v8h*)(q + 16);
  v16h f;
#pragma unroll
  for (int j = 0; j < 8; ++j) { f[j] = lo[j]; f[j + 8] = hi[j]; }
  return f;
}

// B (32x16): source stored as B^T row-major (N x K): lane holds column
// n=l&15; lanes 0-15 K={0..7,16..23}, lanes 16-31 K={8..15,24..31}.
static __device__ __forceinline__ v16h fragB_f16(const _Float16* p, int ld, int lane) {
  int col = lane & 15;
  int ko  = (lane >> 4) << 3;
  const _Float16* q = p + col * ld + ko;
  v8h lo = *(const v8h*)(q);
  v8h hi = *(const v8h*)(q + 16);
  v16h f;
#pragma unroll
  for (int j = 0; j < 8; ++j) { f[j] = lo[j]; f[j + 8] = hi[j]; }
  return f;
}

static __device__ __forceinline__ v8f wmma_f16(v16h a, v16h b, v8f c) {
  return __builtin_amdgcn_wmma_f32_16x16x32_f16(false, a, false, b, (short)0, c,
                                                false, false);
}

// ---------------- Kernel 0: f32 -> f16 weight conversion -------------------

__global__ void cvt_kernel(const float* __restrict__ src,
                           _Float16* __restrict__ dst, int n) {
  int i = (blockIdx.x * 256 + threadIdx.x) * 4;
  if (i < n) {
    v4f v = *(const v4f*)(src + i);
    v2h a = { (_Float16)v[0], (_Float16)v[1] };
    v2h b = { (_Float16)v[2], (_Float16)v[3] };
    *(v2h*)(dst + i)     = a;
    *(v2h*)(dst + i + 2) = b;
  }
}

// ---------------- Kernel 1: GroupNorm -> xn_t (B, T, C) f16 ----------------

__global__ void gn_kernel(const float* __restrict__ x,
                          const float* __restrict__ scale,
                          const float* __restrict__ bias,
                          _Float16* __restrict__ xn_t) {
  int b = blockIdx.x >> 5;
  int g = blockIdx.x & 31;
  const float* xg = x + ((size_t)b * NC + g * 16) * NT;  // 16 ch x 1024

  float s = 0.f, s2 = 0.f;
  for (int i = threadIdx.x; i < 16 * NT; i += 256) {
    float v = xg[i];
    s += v; s2 += v * v;
  }
  __shared__ float rs[256], rq[256];
  rs[threadIdx.x] = s; rq[threadIdx.x] = s2;
  __syncthreads();
  for (int off = 128; off > 0; off >>= 1) {
    if (threadIdx.x < off) {
      rs[threadIdx.x] += rs[threadIdx.x + off];
      rq[threadIdx.x] += rq[threadIdx.x + off];
    }
    __syncthreads();
  }
  float mean = rs[0] * (1.f / 16384.f);
  float var  = rq[0] * (1.f / 16384.f) - mean * mean;
  float rstd = rsqrtf(var + 1e-5f);

  for (int i = threadIdx.x; i < 16 * NT; i += 256) {
    int c  = i >> 10;
    int t  = i & (NT - 1);
    int cg = g * 16 + c;
    float v = (xg[i] - mean) * rstd * scale[cg] + bias[cg];
    xn_t[((size_t)b * NT + t) * NC + cg] = (_Float16)v;
  }
}

// ---------------- Kernel 2: QKV GEMM (16x128 per wave) ---------------------
// out[b,o,t] = sum_k w_qkv[o,k]*xn[b,k,t] + b_qkv[o]
// o -> head=o/192, r=o%192; r<64: q, <128: k, else v.
// q_t,k_t: (bh,T,ch) f16 pre-scaled; v: (bh,ch,T) f16.

__global__ void qkv_kernel(const _Float16* __restrict__ wq_h,
                           const float* __restrict__ b_qkv,
                           const _Float16* __restrict__ xn_t,
                           _Float16* __restrict__ q_t,
                           _Float16* __restrict__ k_t,
                           _Float16* __restrict__ v_n) {
  int lane = threadIdx.x & 31;
  int wid  = blockIdx.x * 8 + (threadIdx.x >> 5);   // 6144 wave-tasks
  int ng  = wid & 7;            // n-group: 128 columns
  int tmp = wid >> 3;
  int mo  = tmp % 96;
  int b   = tmp / 96;
  int m0 = mo * 16, t0 = ng * 128;

  const _Float16* xb = xn_t + (size_t)b * NT * NC + (size_t)t0 * NC;
  const _Float16* wb = wq_h + (size_t)m0 * NC;

  v8f acc[8];
#pragma unroll
  for (int j = 0; j < 8; ++j) acc[j] = (v8f){};

  for (int ks = 0; ks < 16; ++ks) {
    if (ks < 15) __builtin_prefetch(xb + (ks + 1) * 32, 0, 1);
    v16h a = fragA_f16(wb + ks * 32, NC, lane);
#pragma unroll
    for (int j = 0; j < 8; ++j) {
      v16h bb = fragB_f16(xb + (size_t)(j * 16) * NC + ks * 32, NC, lane);
      acc[j] = wmma_f16(a, bb, acc[j]);
    }
  }

  int head = m0 / 192;
  int r0   = m0 % 192;
  int sec  = r0 >> 6;          // 0=q 1=k 2=v (tile never straddles sections)
  int c0   = r0 & 63;
  int bh   = b * NHD + head;
  int n  = lane & 15;
  int mh = (lane >> 4) << 3;

#pragma unroll
  for (int j = 0; j < 8; ++j) {
    int t = t0 + j * 16 + n;
    if (sec == 2) {
#pragma unroll
      for (int r = 0; r < 8; ++r) {
        int m = mh + r;
        float val = acc[j][r] + b_qkv[m0 + m];
        v_n[(size_t)bh * NCH * NT + (size_t)(c0 + m) * NT + t] = (_Float16)val;
      }
    } else {
      _Float16* dst = (sec == 0 ? q_t : k_t) +
                      (size_t)bh * NT * NCH + (size_t)t * NCH + c0 + mh;
#pragma unroll
      for (int r = 0; r < 8; r += 2) {
        float v0 = (acc[j][r]     + b_qkv[m0 + mh + r])     * QK_SCALE;
        float v1 = (acc[j][r + 1] + b_qkv[m0 + mh + r + 1]) * QK_SCALE;
        v2h pk = { (_Float16)v0, (_Float16)v1 };
        *(v2h*)(dst + r) = pk;
      }
    }
  }
}

// ---------------- Kernel 3: attention ---------------------------------------
// 128 threads = 4 waves; each wave owns 16 query rows and a 16x1024 f16
// strip of exp(scores) in LDS. 1/sum folded into the P*V^T epilogue.

__global__ void attn_kernel(const _Float16* __restrict__ q_t,
                            const _Float16* __restrict__ k_t,
                            const _Float16* __restrict__ v_n,
                            _Float16* __restrict__ h_t) {
  extern __shared__ char smem_raw[];
  _Float16* Sall = (_Float16*)smem_raw;                       // 4*16*1024 f16
  float*    inv  = (float*)(smem_raw + 4 * 16 * NT * sizeof(_Float16));

  int bh   = blockIdx.x >> 4;
  int tb   = blockIdx.x & 15;
  int w    = threadIdx.x >> 5;
  int lane = threadIdx.x & 31;
  int t0   = tb * 64 + w * 16;

  _Float16* S = Sall + (size_t)w * 16 * NT;

  const _Float16* qb = q_t + (size_t)bh * NT * NCH + (size_t)t0 * NCH;
  const _Float16* kb = k_t + (size_t)bh * NT * NCH;
  const _Float16* vb = v_n + (size_t)bh * NCH * NT;

  // ---- raw scores -> f16 LDS strip: S[16 x 1024] = q^T k (K = ch = 64) ----
  v16h a0 = fragA_f16(qb, NCH, lane);
  v16h a1 = fragA_f16(qb + 32, NCH, lane);
  int n  = lane & 15;
  int mh = (lane >> 4) << 3;
#pragma unroll 2
  for (int st = 0; st < 64; ++st) {
    const _Float16* kp = kb + (size_t)(st * 16) * NCH;
    v8f acc = {};
    acc = wmma_f16(a0, fragB_f16(kp, NCH, lane), acc);
    acc = wmma_f16(a1, fragB_f16(kp + 32, NCH, lane), acc);
#pragma unroll
    for (int r = 0; r < 8; ++r)
      S[(size_t)(mh + r) * NT + st * 16 + n] = (_Float16)acc[r];
  }
  __syncthreads();

  // ---- wave-local softmax: S <- exp(S - max); keep 1/sum per row ----
  for (int row = 0; row < 16; ++row) {
    _Float16* Sr = S + (size_t)row * NT;
    float m = -3.4e38f;
    for (int i = lane; i < NT; i += 32) m = fmaxf(m, (float)Sr[i]);
#pragma unroll
    for (int off = 16; off > 0; off >>= 1) m = fmaxf(m, __shfl_xor(m, off));
    float sum = 0.f;
    for (int i = lane; i < NT; i += 32) {
      float e = __expf((float)Sr[i] - m);
      Sr[i] = (_Float16)e;
      sum += e;
    }
#pragma unroll
    for (int off = 16; off > 0; off >>= 1) sum += __shfl_xor(sum, off);
    if (lane == 0) inv[w * 16 + row] = 1.0f / sum;
  }
  __syncthreads();

  // ---- O[16x64] = P(16x1024) x V^T(1024x64); P frag reused over c-tiles ---
  int b    = bh >> 3;
  int head = bh & 7;
  v8f acc[4];
#pragma unroll
  for (int ct = 0; ct < 4; ++ct) acc[ct] = (v8f){};

  for (int ks = 0; ks < 32; ++ks) {
    v16h aP = fragA_f16(S + ks * 32, NT, lane);
#pragma unroll
    for (int ct = 0; ct < 4; ++ct) {
      v16h bV = fragB_f16(vb + (size_t)(ct * 16) * NT + ks * 32, NT, lane);
      acc[ct] = wmma_f16(aP, bV, acc[ct]);
    }
  }

#pragma unroll
  for (int ct = 0; ct < 4; ++ct) {
    int cg = head * NCH + ct * 16 + n;
#pragma unroll
    for (int r = 0; r < 8; ++r) {
      float o = acc[ct][r] * inv[w * 16 + mh + r];   // fold softmax 1/sum here
      h_t[((size_t)b * NT + t0 + mh + r) * NC + cg] = (_Float16)o;
    }
  }
}

// ---------------- Kernel 4: proj GEMM (16x128 per wave) + residual ---------

__global__ void proj_kernel(const _Float16* __restrict__ wp_h,
                            const float* __restrict__ b_proj,
                            const _Float16* __restrict__ h_t,
                            const float* __restrict__ x,
                            float* __restrict__ out) {
  int lane = threadIdx.x & 31;
  int wid  = blockIdx.x * 8 + (threadIdx.x >> 5);   // 2048 wave-tasks
  int ng  = wid & 7;
  int tmp = wid >> 3;
  int mo  = tmp & 31;
  int b   = tmp >> 5;
  int m0 = mo * 16, t0 = ng * 128;

  const _Float16* hb = h_t + (size_t)b * NT * NC + (size_t)t0 * NC;
  const _Float16* wb = wp_h + (size_t)m0 * NC;

  v8f acc[8];
#pragma unroll
  for (int j = 0; j < 8; ++j) acc[j] = (v8f){};

  for (int ks = 0; ks < 16; ++ks) {
    if (ks < 15) __builtin_prefetch(hb + (ks + 1) * 32, 0, 1);
    v16h a = fragA_f16(wb + ks * 32, NC, lane);
#pragma unroll
    for (int j = 0; j < 8; ++j) {
      v16h bb = fragB_f16(hb + (size_t)(j * 16) * NC + ks * 32, NC, lane);
      acc[j] = wmma_f16(a, bb, acc[j]);
    }
  }

  int n  = lane & 15;
  int mh = (lane >> 4) << 3;
#pragma unroll
  for (int j = 0; j < 8; ++j) {
    int t = t0 + j * 16 + n;
#pragma unroll
    for (int r = 0; r < 8; ++r) {
      int m = mh + r;
      size_t idx = ((size_t)b * NC + m0 + m) * NT + t;
      out[idx] = acc[j][r] + b_proj[m0 + m] + x[idx];
    }
  }
}

// ---------------- Launch ----------------------------------------------------

extern "C" void kernel_launch(void* const* d_in, const int* in_sizes, int n_in,
                              void* d_out, int out_size, void* d_ws, size_t ws_size,
                              hipStream_t stream) {
  const float* x        = (const float*)d_in[0];
  const float* gn_scale = (const float*)d_in[1];
  const float* gn_bias  = (const float*)d_in[2];
  const float* w_qkv    = (const float*)d_in[3];
  const float* b_qkv    = (const float*)d_in[4];
  const float* w_proj   = (const float*)d_in[5];
  const float* b_proj   = (const float*)d_in[6];
  float* out = (float*)d_out;

  char* ws = (char*)d_ws;
  const size_t SZ = (size_t)NB * NT * NC * sizeof(_Float16);   // 8 MB each
  _Float16* xn_t = (_Float16*)(ws + 0 * SZ);   // (B, T, C)
  _Float16* q_t  = (_Float16*)(ws + 1 * SZ);   // (BH, T, ch)
  _Float16* k_t  = (_Float16*)(ws + 2 * SZ);   // (BH, T, ch)
  _Float16* v_n  = (_Float16*)(ws + 3 * SZ);   // (BH, ch, T)
  _Float16* h_t  = (_Float16*)(ws + 4 * SZ);   // (B, T, C)
  _Float16* wq_h = (_Float16*)(ws + 5 * SZ);                    // 1536x512 f16
  _Float16* wp_h = (_Float16*)(ws + 5 * SZ + (size_t)3 * NC * NC * 2);

  cvt_kernel<<<(3 * NC * NC) / 1024, 256, 0, stream>>>(w_qkv, wq_h, 3 * NC * NC);
  cvt_kernel<<<(NC * NC) / 1024, 256, 0, stream>>>(w_proj, wp_h, NC * NC);

  gn_kernel<<<NB * 32, 256, 0, stream>>>(x, gn_scale, gn_bias, xn_t);

  // 8 b * 96 mo * 8 ng = 6144 wave-tasks / 8 waves per block
  qkv_kernel<<<6144 / 8, 256, 0, stream>>>(wq_h, b_qkv, xn_t, q_t, k_t, v_n);

  // 64 bh * 16 query-blocks; 4 waves/block; 128KB P + 256B inv in LDS
  size_t attn_lds = (size_t)4 * 16 * NT * sizeof(_Float16) + 64 * sizeof(float);
  attn_kernel<<<NBH * 16, 128, attn_lds, stream>>>(q_t, k_t, v_n, h_t);

  // 8 b * 32 mo * 8 ng = 2048 wave-tasks / 8 waves per block
  proj_kernel<<<2048 / 8, 256, 0, stream>>>(wp_h, b_proj, h_t, x, out);
}
